// MultiHeadAttention_50036368998915
// MI455X (gfx1250) — matrix operand order repaired and checked by reading, not compiled, not run on hardware
//
#include <hip/hip_runtime.h>
#include <hip/hip_bf16.h>

// ---------------------------------------------------------------------------
// MHA for MI455X (gfx1250): bf16 WMMA everywhere, flash-attention inner loop
// with double-buffered GLOBAL_LOAD_ASYNC_TO_LDS staging of K/V tiles.
// ---------------------------------------------------------------------------

#define EMBED    1024
#define HEADS    16
#define HEAD_DIM 64
#define BATCH    2
#define SEQ      4096
#define M_TOTAL  (BATCH * SEQ)   // 8192 rows for the projection GEMMs

typedef __attribute__((ext_vector_type(16))) __bf16 v16bf;
typedef __attribute__((ext_vector_type(8)))  __bf16 v8bf;
typedef __attribute__((ext_vector_type(8)))  float  v8f;

union ABFrag { v16bf v; v8bf h[2]; };

__device__ __forceinline__ unsigned short f32_to_bf16(float f) {
  union { float f; unsigned int u; } cv; cv.f = f;
  unsigned int u = cv.u;
  u += 0x7FFFu + ((u >> 16) & 1u);   // round-to-nearest-even
  return (unsigned short)(u >> 16);
}

__device__ __forceinline__ v8f wmma_bf16(const ABFrag& a, const ABFrag& b, v8f c) {
  return __builtin_amdgcn_wmma_f32_16x16x32_bf16(
      /*neg_a=*/false, a.v, /*neg_b=*/false, b.v,
      /*c_mod=*/(short)0, c, /*reuse_a=*/false, /*reuse_b=*/false);
}

// Async copy 16B global -> LDS (ASYNCcnt-tracked).  lds_off is the workgroup
// LDS byte offset (low 32 bits of a flat shared-memory address).
__device__ __forceinline__ void async_copy_b128(unsigned lds_off, const void* gptr) {
  asm volatile("global_load_async_to_lds_b128 %0, %1, off"
               :
               : "v"(lds_off), "v"(gptr)
               : "memory");
}

// ---------------------------------------------------------------------------
// Kernel 1: fp32 -> bf16 cast (grid-stride)
// ---------------------------------------------------------------------------
__global__ void cast_f32_bf16(const float* __restrict__ src,
                              unsigned short* __restrict__ dst, int n) {
  int i = blockIdx.x * blockDim.x + threadIdx.x;
  int stride = gridDim.x * blockDim.x;
  for (; i < n; i += stride) dst[i] = f32_to_bf16(src[i]);
}

// ---------------------------------------------------------------------------
// Kernel 2: fused QKV projection.  Y = x @ W.T   (x: [M,1024] bf16, W: [N,K])
// Block = 128 threads = 4 waves, each wave a 32x32 C tile; block covers 64x64.
// Q,K written as [b,h,s,d]; V written transposed [b,h,d,s].
// ---------------------------------------------------------------------------
__global__ __launch_bounds__(128)
void qkv_proj(const unsigned short* __restrict__ xb,
              const unsigned short* __restrict__ wq,
              const unsigned short* __restrict__ wk,
              const unsigned short* __restrict__ wv,
              unsigned short* __restrict__ qout,
              unsigned short* __restrict__ kout,
              unsigned short* __restrict__ vtout) {
  const int lane = threadIdx.x & 31;
  const int wave = threadIdx.x >> 5;
  const int hl   = lane >> 4;        // half of wave (0/1)
  const int l16  = lane & 15;
  const int m0 = blockIdx.x * 64 + (wave >> 1) * 32;
  const int n0 = blockIdx.y * 64 + (wave & 1) * 32;
  const int which = blockIdx.z;
  const unsigned short* w = (which == 0) ? wq : ((which == 1) ? wk : wv);

  v8f c00 = {}, c01 = {}, c10 = {}, c11 = {};

  // A-frag rows: m = m0(+16) + l16; e -> k = k0 + hl*16 + e (16 consecutive)
  const unsigned short* aRow0 = xb + (size_t)(m0 + l16) * EMBED + hl * 16;
  const unsigned short* aRow1 = aRow0 + (size_t)16 * EMBED;
  // B-frag: n = n0(+16) + l16; e -> k = k0 + hl*16 + e (16 consecutive)
  const unsigned short* bRow0 = w + (size_t)(n0 + l16) * EMBED + hl * 16;
  const unsigned short* bRow1 = bRow0 + (size_t)16 * EMBED;

  for (int k0 = 0; k0 < EMBED; k0 += 32) {
    ABFrag a0, a1, b0, b1;
    a0.h[0] = *(const v8bf*)(aRow0 + k0);
    a0.h[1] = *(const v8bf*)(aRow0 + k0 + 8);
    a1.h[0] = *(const v8bf*)(aRow1 + k0);
    a1.h[1] = *(const v8bf*)(aRow1 + k0 + 8);
    b0.h[0] = *(const v8bf*)(bRow0 + k0);
    b0.h[1] = *(const v8bf*)(bRow0 + k0 + 8);
    b1.h[0] = *(const v8bf*)(bRow1 + k0);
    b1.h[1] = *(const v8bf*)(bRow1 + k0 + 8);
    c00 = wmma_bf16(a0, b0, c00);
    c01 = wmma_bf16(a0, b1, c01);
    c10 = wmma_bf16(a1, b0, c10);
    c11 = wmma_bf16(a1, b1, c11);
  }

  auto storeQK = [&](unsigned short* dst, const v8f& c, int mb, int nb) {
#pragma unroll
    for (int r = 0; r < 8; ++r) {
      int m = mb + hl * 8 + r;
      int n = nb + l16;
      int b = m / SEQ, s = m % SEQ;
      int h = n / HEAD_DIM, d = n % HEAD_DIM;
      dst[((size_t)(b * HEADS + h) * SEQ + s) * HEAD_DIM + d] = f32_to_bf16(c[r]);
    }
  };
  auto storeVT = [&](unsigned short* dst, const v8f& c, int mb, int nb) {
#pragma unroll
    for (int r = 0; r < 8; ++r) {
      int m = mb + hl * 8 + r;
      int n = nb + l16;
      int b = m / SEQ, s = m % SEQ;
      int h = n / HEAD_DIM, d = n % HEAD_DIM;
      dst[((size_t)(b * HEADS + h) * HEAD_DIM + d) * SEQ + s] = f32_to_bf16(c[r]);
    }
  };

  if (which == 2) {
    storeVT(vtout, c00, m0, n0);       storeVT(vtout, c01, m0, n0 + 16);
    storeVT(vtout, c10, m0 + 16, n0);  storeVT(vtout, c11, m0 + 16, n0 + 16);
  } else {
    unsigned short* dst = (which == 0) ? qout : kout;
    storeQK(dst, c00, m0, n0);         storeQK(dst, c01, m0, n0 + 16);
    storeQK(dst, c10, m0 + 16, n0);    storeQK(dst, c11, m0 + 16, n0 + 16);
  }
}

// ---------------------------------------------------------------------------
// Kernel 3: flash attention.  One wave = 16 query rows of one (b,h).
// Block = 4 waves = 64 queries.  Key loop in tiles of 32 keys, with the K/V
// tiles double-buffered in LDS via GLOBAL_LOAD_ASYNC_TO_LDS_B128.
// ---------------------------------------------------------------------------
__global__ __launch_bounds__(128)
void flash_attn(const unsigned short* __restrict__ Qm,
                const unsigned short* __restrict__ Km,
                const unsigned short* __restrict__ Vt,
                unsigned short* __restrict__ attn) {
  __shared__ __align__(16) unsigned short kTile[2][32 * HEAD_DIM]; // [key][d]
  __shared__ __align__(16) unsigned short vTile[2][HEAD_DIM * 32]; // [d][key]
  __shared__ __align__(16) unsigned short plds[4][16 * 32];

  const int lane = threadIdx.x & 31;
  const int wave = threadIdx.x >> 5;
  const int hl   = lane >> 4;
  const int l16  = lane & 15;
  const int tid  = threadIdx.x;

  const int bh = blockIdx.x;                    // 0..31  (b*HEADS + h)
  const int q0 = blockIdx.y * 64 + wave * 16;   // query tile base

  const unsigned short* Qb = Qm + (size_t)bh * SEQ * HEAD_DIM;
  const unsigned short* Kb = Km + (size_t)bh * SEQ * HEAD_DIM;
  const unsigned short* Vb = Vt + (size_t)bh * HEAD_DIM * SEQ;

  // Stage the K tile (32 keys x 64 d) and V tile (64 d x 32 keys) for key
  // base `ktN` into LDS buffer `bufIdx`: 512 x 16B chunks, 4 per thread.
  auto stage = [&](int ktN, int bufIdx) {
#pragma unroll
    for (int c = tid; c < 256; c += 128) {       // K tile: 8 chunks / 128B row
      int key = c >> 3;
      int dB  = (c & 7) * 16;
      unsigned ldsOff = (unsigned)(size_t)(&kTile[bufIdx][0]) + c * 16;
      const char* g = (const char*)(Kb + (size_t)(ktN + key) * HEAD_DIM) + dB;
      async_copy_b128(ldsOff, g);
    }
#pragma unroll
    for (int c = tid; c < 256; c += 128) {       // V tile: 4 chunks / 64B row
      int d  = c >> 2;
      int kB = (c & 3) * 16;
      unsigned ldsOff = (unsigned)(size_t)(&vTile[bufIdx][0]) + c * 16;
      const char* g = (const char*)(Vb + (size_t)d * SEQ + ktN) + kB;
      async_copy_b128(ldsOff, g);
    }
  };

  // Q A-fragments (d 0..31 and 32..63), loaded once.
  ABFrag qa0, qa1;
  {
    const unsigned short* qrow = Qb + (size_t)(q0 + l16) * HEAD_DIM + hl * 16;
    qa0.h[0] = *(const v8bf*)(qrow + 0);
    qa0.h[1] = *(const v8bf*)(qrow + 8);
    qa1.h[0] = *(const v8bf*)(qrow + 32);
    qa1.h[1] = *(const v8bf*)(qrow + 40);
  }

  v8f o0 = {}, o1 = {}, o2 = {}, o3 = {};      // output accs, d tiles 0..3
  float mrow[8], lrow[8];
#pragma unroll
  for (int r = 0; r < 8; ++r) { mrow[r] = -3.0e38f; lrow[r] = 0.f; }

  unsigned short* myP = plds[wave];
  const float scale = 0.125f;                  // 1/sqrt(64)

  stage(0, 0);                                 // prologue: tile 0 in flight

  for (int kt = 0; kt < SEQ; kt += 32) {
    const int buf = (kt >> 5) & 1;
    if (kt + 32 < SEQ) {
      stage(kt + 32, buf ^ 1);                 // prefetch next tile
      asm volatile("s_wait_asynccnt 0x4" ::: "memory");  // current tile done
    } else {
      asm volatile("s_wait_asynccnt 0x0" ::: "memory");
    }
    __syncthreads();                           // all waves' tiles visible

    // ---- scores S[16 x 32] = Q @ K^T (B-frags from LDS) ------------------
    v8f s0 = {}, s1 = {};
    const unsigned short* kr0 = &kTile[buf][(size_t)l16 * HEAD_DIM + hl * 16];
    const unsigned short* kr1 = kr0 + 16 * HEAD_DIM;
    ABFrag b00, b10, b01, b11;
    b00.h[0] = *(const v8bf*)(kr0);        b00.h[1] = *(const v8bf*)(kr0 + 8);   // d 0..31
    b10.h[0] = *(const v8bf*)(kr0 + 32);   b10.h[1] = *(const v8bf*)(kr0 + 40);  // d 32..63
    b01.h[0] = *(const v8bf*)(kr1);        b01.h[1] = *(const v8bf*)(kr1 + 8);
    b11.h[0] = *(const v8bf*)(kr1 + 32);   b11.h[1] = *(const v8bf*)(kr1 + 40);
    s0 = wmma_bf16(qa0, b00, s0);  s0 = wmma_bf16(qa1, b10, s0);
    s1 = wmma_bf16(qa0, b01, s1);  s1 = wmma_bf16(qa1, b11, s1);

    // ---- online softmax: rows live across each 16-lane group -------------
    float p0[8], p1[8], alpha[8];
#pragma unroll
    for (int r = 0; r < 8; ++r) {
      float a = s0[r] * scale, b = s1[r] * scale;
      float t = fmaxf(a, b);
#pragma unroll
      for (int off = 8; off; off >>= 1) t = fmaxf(t, __shfl_xor(t, off, 16));
      float mnew = fmaxf(mrow[r], t);
      float al = __expf(mrow[r] - mnew);
      mrow[r] = mnew; alpha[r] = al;
      float e0 = __expf(a - mnew), e1 = __expf(b - mnew);
      float ps = e0 + e1;
#pragma unroll
      for (int off = 8; off; off >>= 1) ps += __shfl_xor(ps, off, 16);
      lrow[r] = lrow[r] * al + ps;
      p0[r] = e0; p1[r] = e1;
    }
#pragma unroll
    for (int r = 0; r < 8; ++r) {
      o0[r] *= alpha[r]; o1[r] *= alpha[r]; o2[r] *= alpha[r]; o3[r] *= alpha[r];
    }

    // ---- C-layout -> A-layout reshuffle of P via per-wave LDS tile -------
#pragma unroll
    for (int r = 0; r < 8; ++r) {
      int row = hl * 8 + r;
      myP[row * 32 + l16]      = f32_to_bf16(p0[r]);
      myP[row * 32 + l16 + 16] = f32_to_bf16(p1[r]);
    }
    asm volatile("s_wait_dscnt 0x0" ::: "memory");  // wave-local DS fence
    ABFrag pf;
    {
      const unsigned short* pr = myP + l16 * 32 + hl * 16;
      pf.h[0] = *(const v8bf*)(pr);
      pf.h[1] = *(const v8bf*)(pr + 8);
    }

    // ---- O += P @ V (V tile [d][key] in LDS -> contiguous B-frags) -------
    const unsigned short* vcol = &vTile[buf][(size_t)l16 * 32 + hl * 16];
    ABFrag vb0, vb1, vb2, vb3;
    vb0.h[0] = *(const v8bf*)(vcol);             vb0.h[1] = *(const v8bf*)(vcol + 8);
    vb1.h[0] = *(const v8bf*)(vcol + 16 * 32);   vb1.h[1] = *(const v8bf*)(vcol + 16 * 32 + 8);
    vb2.h[0] = *(const v8bf*)(vcol + 32 * 32);   vb2.h[1] = *(const v8bf*)(vcol + 32 * 32 + 8);
    vb3.h[0] = *(const v8bf*)(vcol + 48 * 32);   vb3.h[1] = *(const v8bf*)(vcol + 48 * 32 + 8);
    o0 = wmma_bf16(pf, vb0, o0);
    o1 = wmma_bf16(pf, vb1, o1);
    o2 = wmma_bf16(pf, vb2, o2);
    o3 = wmma_bf16(pf, vb3, o3);

    __syncthreads();   // compute done before next prefetch overwrites buf^1
  }

  // ---- finalize: divide by row sums, write bf16 attn [b, s, e] ----------
  const int b = bh / HEADS, h = bh % HEADS;
#pragma unroll
  for (int r = 0; r < 8; ++r) {
    int row = q0 + hl * 8 + r;
    float inv = 1.0f / lrow[r];
    size_t base = ((size_t)b * SEQ + row) * EMBED + h * HEAD_DIM;
    attn[base +  0 + l16] = f32_to_bf16(o0[r] * inv);
    attn[base + 16 + l16] = f32_to_bf16(o1[r] * inv);
    attn[base + 32 + l16] = f32_to_bf16(o2[r] * inv);
    attn[base + 48 + l16] = f32_to_bf16(o3[r] * inv);
  }
}

// ---------------------------------------------------------------------------
// Kernel 4: output projection  out = attn @ Wo.T + bo   (f32 output)
// ---------------------------------------------------------------------------
__global__ __launch_bounds__(128)
void out_proj(const unsigned short* __restrict__ ab,
              const unsigned short* __restrict__ wo,
              const float* __restrict__ bo,
              float* __restrict__ out) {
  const int lane = threadIdx.x & 31;
  const int wave = threadIdx.x >> 5;
  const int hl   = lane >> 4;
  const int l16  = lane & 15;
  const int m0 = blockIdx.x * 64 + (wave >> 1) * 32;
  const int n0 = blockIdx.y * 64 + (wave & 1) * 32;

  v8f c00 = {}, c01 = {}, c10 = {}, c11 = {};

  const unsigned short* aRow0 = ab + (size_t)(m0 + l16) * EMBED + hl * 16;
  const unsigned short* aRow1 = aRow0 + (size_t)16 * EMBED;
  const unsigned short* bRow0 = wo + (size_t)(n0 + l16) * EMBED + hl * 16;
  const unsigned short* bRow1 = bRow0 + (size_t)16 * EMBED;

  for (int k0 = 0; k0 < EMBED; k0 += 32) {
    ABFrag a0, a1, b0, b1;
    a0.h[0] = *(const v8bf*)(aRow0 + k0);
    a0.h[1] = *(const v8bf*)(aRow0 + k0 + 8);
    a1.h[0] = *(const v8bf*)(aRow1 + k0);
    a1.h[1] = *(const v8bf*)(aRow1 + k0 + 8);
    b0.h[0] = *(const v8bf*)(bRow0 + k0);
    b0.h[1] = *(const v8bf*)(bRow0 + k0 + 8);
    b1.h[0] = *(const v8bf*)(bRow1 + k0);
    b1.h[1] = *(const v8bf*)(bRow1 + k0 + 8);
    c00 = wmma_bf16(a0, b0, c00);
    c01 = wmma_bf16(a0, b1, c01);
    c10 = wmma_bf16(a1, b0, c10);
    c11 = wmma_bf16(a1, b1, c11);
  }

  float bias0 = bo[n0 + l16];
  float bias1 = bo[n0 + 16 + l16];
#pragma unroll
  for (int r = 0; r < 8; ++r) {
    int mA = m0 + hl * 8 + r;
    int mB = mA + 16;
    out[(size_t)mA * EMBED + n0 + l16]      = c00[r] + bias0;
    out[(size_t)mA * EMBED + n0 + 16 + l16] = c01[r] + bias1;
    out[(size_t)mB * EMBED + n0 + l16]      = c10[r] + bias0;
    out[(size_t)mB * EMBED + n0 + 16 + l16] = c11[r] + bias1;
  }
}

// ---------------------------------------------------------------------------
// Workspace layout (bytes)
// ---------------------------------------------------------------------------
#define OFF_XB   ((size_t)0)                         // x bf16       16 MiB
#define OFF_WQ   (OFF_XB  + (size_t)M_TOTAL * EMBED * 2)
#define OFF_WK   (OFF_WQ  + (size_t)EMBED * EMBED * 2)
#define OFF_WV   (OFF_WK  + (size_t)EMBED * EMBED * 2)
#define OFF_WO   (OFF_WV  + (size_t)EMBED * EMBED * 2)
#define OFF_Q    (OFF_WO  + (size_t)EMBED * EMBED * 2)
#define OFF_K    (OFF_Q   + (size_t)M_TOTAL * EMBED * 2)
#define OFF_VT   (OFF_K   + (size_t)M_TOTAL * EMBED * 2)
#define OFF_ATTN (OFF_VT  + (size_t)M_TOTAL * EMBED * 2)

extern "C" void kernel_launch(void* const* d_in, const int* in_sizes, int n_in,
                              void* d_out, int out_size, void* d_ws, size_t ws_size,
                              hipStream_t stream) {
  const float* x  = (const float*)d_in[0];
  const float* Wq = (const float*)d_in[1];
  const float* Wk = (const float*)d_in[2];
  const float* Wv = (const float*)d_in[3];
  const float* Wo = (const float*)d_in[4];
  const float* bo = (const float*)d_in[5];
  float* out = (float*)d_out;

  char* ws = (char*)d_ws;
  unsigned short* xb  = (unsigned short*)(ws + OFF_XB);
  unsigned short* wqb = (unsigned short*)(ws + OFF_WQ);
  unsigned short* wkb = (unsigned short*)(ws + OFF_WK);
  unsigned short* wvb = (unsigned short*)(ws + OFF_WV);
  unsigned short* wob = (unsigned short*)(ws + OFF_WO);
  unsigned short* qb  = (unsigned short*)(ws + OFF_Q);
  unsigned short* kb  = (unsigned short*)(ws + OFF_K);
  unsigned short* vtb = (unsigned short*)(ws + OFF_VT);
  unsigned short* atb = (unsigned short*)(ws + OFF_ATTN);

  // 1) casts
  cast_f32_bf16<<<2048, 256, 0, stream>>>(x,  xb,  M_TOTAL * EMBED);
  cast_f32_bf16<<<512,  256, 0, stream>>>(Wq, wqb, EMBED * EMBED);
  cast_f32_bf16<<<512,  256, 0, stream>>>(Wk, wkb, EMBED * EMBED);
  cast_f32_bf16<<<512,  256, 0, stream>>>(Wv, wvb, EMBED * EMBED);
  cast_f32_bf16<<<512,  256, 0, stream>>>(Wo, wob, EMBED * EMBED);

  // 2) fused QKV projection: grid = (M/64, N/64, 3)
  qkv_proj<<<dim3(M_TOTAL / 64, EMBED / 64, 3), 128, 0, stream>>>(
      xb, wqb, wkb, wvb, qb, kb, vtb);

  // 3) flash attention: grid = (B*H, S/64)
  flash_attn<<<dim3(BATCH * HEADS, SEQ / 64), 128, 0, stream>>>(qb, kb, vtb, atb);

  // 4) output projection + bias
  out_proj<<<dim3(M_TOTAL / 64, EMBED / 64), 128, 0, stream>>>(atb, wob, bo, out);

  (void)in_sizes; (void)n_in; (void)out_size; (void)ws_size;
}
